// CustomMultiHeadAttention_24541443129418
// MI455X (gfx1250) — matrix operand order repaired and checked by reading, not compiled, not run on hardware
//
#include <hip/hip_runtime.h>

// ---------------------------------------------------------------------------
// MI455X (gfx1250, wave32) fused multi-head attention:
//   elu(X@W+b) projections (WMMA bf16) -> flash attention (WMMA bf16, online
//   softmax, async double-buffered K tiles, logits never hit HBM)
//   -> elu(ctx@Wo+bo) (WMMA bf16, fp32 out).
// ---------------------------------------------------------------------------

#define BB    4
#define SS    2048
#define DDIM  1024
#define HH    16
#define DEPTH 64

typedef __attribute__((ext_vector_type(16))) __bf16 v16bf;
typedef __attribute__((ext_vector_type(8)))  __bf16 v8bf;
typedef __attribute__((ext_vector_type(8)))  float  v8f;

union Frag { v16bf v; v8bf h[2]; };

__device__ __forceinline__ __bf16 f2bf(float f) {
  unsigned u = __builtin_bit_cast(unsigned, f);
  unsigned r = u + 0x7fffu + ((u >> 16) & 1u);   // round-to-nearest-even
  unsigned short hv = (unsigned short)(r >> 16);
  return __builtin_bit_cast(__bf16, hv);
}

// Async global->LDS copy, 16B per lane (ASYNCcnt-tracked, no VGPR data path).
// ldsOff: LDS byte address (generic shared ptr truncated to 32b per aperture).
__device__ __forceinline__ void async_b128(const void* ldsPtr, const void* g) {
  unsigned ldsOff = (unsigned)(unsigned long long)ldsPtr;
  asm volatile("global_load_async_to_lds_b128 %0, %1, off"
               :: "v"(ldsOff), "v"(g) : "memory");
}
#define WAIT_ASYNC(n) asm volatile("s_wait_asynccnt " #n ::: "memory")

// A-fragment (16x32 bf16, MxK): lane L: m=L&15, half=L>>4.
// elements 0..7  -> K = kofs + half*8 + i        (contiguous 16B)
// elements 8..15 -> K = kofs + 16 + half*8 + i   (contiguous 16B)
__device__ __forceinline__ v16bf load_frag_a(const __bf16* lds, int row0,
                                             int stride, int kofs) {
  int lane = threadIdx.x & 31;
  int m = lane & 15, hf = lane >> 4;
  const __bf16* p = lds + (row0 + m) * stride + kofs + hf * 8;
  Frag f;
  f.h[0] = *(const v8bf*)p;
  f.h[1] = *(const v8bf*)(p + 16);
  return f.v;
}

// B-fragment (32x16 bf16, KxN) from K-contiguous ("transposed") LDS T[n][k]:
// lane L: n=L&15, half=L>>4; elements 0..7 -> K = kofs + half*8 + i,
// elements 8..15 -> K = kofs + 16 + half*8 + i (contiguous 16B each)
__device__ __forceinline__ v16bf load_frag_b(const __bf16* ldsT, int col0,
                                             int stride, int kofs) {
  int lane = threadIdx.x & 31;
  int n = lane & 15, hf = lane >> 4;
  const __bf16* p = ldsT + (col0 + n) * stride + kofs + hf * 8;
  Frag f;
  f.h[0] = *(const v8bf*)p;
  f.h[1] = *(const v8bf*)(p + 16);
  return f.v;
}

__device__ __forceinline__ float elu(float x) {
  return x > 0.f ? x : (__expf(x) - 1.f);
}

// ---------------------------------------------------------------------------
// GEMM + bias + ELU.  C[M,N] = act(A[M,K] @ W[K,N] + bias)
// A_BF16: A is bf16 (ctx; staged via async-to-LDS) vs fp32 (VALU-converted).
// SPLIT_OUT: write bf16 into split-head [B,H,S,depth] layout, else fp32 flat.
// Block: 256 threads (8 waves), 128x128 tile, 32-wide K steps.
// ---------------------------------------------------------------------------
template <bool A_BF16, bool SPLIT_OUT>
__global__ __launch_bounds__(256) void gemm_elu_kernel(
    const void* __restrict__ Ap, const float* __restrict__ W,
    const float* __restrict__ bias, void* __restrict__ outp,
    int M, int K, int N) {
  constexpr int MT = 128, NT = 128, KT = 32, AST = 40;  // stride 40: 16B-aligned frags
  __shared__ __align__(16) __bf16 As[MT * AST];
  __shared__ __align__(16) __bf16 Wt[NT * AST];   // W tile, transposed [n][k]

  const int tid  = threadIdx.x;
  const int wave = tid >> 5;
  const int lane = tid & 31;
  const int nIdx = lane & 15, hf = lane >> 4;
  const int m0 = blockIdx.x * MT;
  const int n0 = blockIdx.y * NT;

  const float*  Af = (const float*)Ap;
  const __bf16* Ab = (const __bf16*)Ap;

  v8f acc[8];
  for (int t = 0; t < 8; t++)
    for (int e = 0; e < 8; e++) acc[t][e] = 0.f;

  for (int k0 = 0; k0 < K; k0 += KT) {
    __syncthreads();
    // Stage A.
    if constexpr (A_BF16) {
      // Raw bf16 copy: async DMA to LDS, 16B chunks (4 per 32-elem row).
      for (int i = tid; i < MT * 4; i += 256) {
        int r = i >> 2, c = (i & 3) * 8;
        async_b128(&As[r * AST + c], Ab + (size_t)(m0 + r) * K + k0 + c);
      }
    } else {
      if (k0 + KT < K)  // prefetch next fp32 A slab into GL2
        __builtin_prefetch(&Af[(size_t)(m0 + (tid >> 1)) * K + k0 + KT], 0, 1);
      for (int i = tid; i < MT * KT; i += 256) {
        int r = i >> 5, c = i & 31;
        As[r * AST + c] = f2bf(Af[(size_t)(m0 + r) * K + k0 + c]);
      }
    }
    // Stage W transposed (fp32 -> bf16 conversion): Wt[n][k] = W[k0+k][n0+n].
    for (int i = tid; i < KT * NT; i += 256) {
      int kk = i >> 7, n = i & 127;
      Wt[n * AST + kk] = f2bf(W[(size_t)(k0 + kk) * N + n0 + n]);
    }
    if constexpr (A_BF16) WAIT_ASYNC(0);
    __syncthreads();

    v16bf a = load_frag_a(As, wave * 16, AST, 0);
    for (int t = 0; t < 8; t++) {
      v16bf b = load_frag_b(Wt, t * 16, AST, 0);
      acc[t] = __builtin_amdgcn_wmma_f32_16x16x32_bf16(
          false, a, false, b, (short)0, acc[t], false, false);
    }
  }

  // Epilogue: bias + ELU + store.  C layout: VGPR r -> row (hf*8+r), col nIdx.
  for (int t = 0; t < 8; t++) {
    int n = n0 + t * 16 + nIdx;
    float bv = bias[n];
    for (int r = 0; r < 8; r++) {
      int m = m0 + wave * 16 + hf * 8 + r;
      float x = elu(acc[t][r] + bv);
      if (SPLIT_OUT) {
        int b = m / SS, s = m % SS, h = n >> 6, d = n & 63;
        ((__bf16*)outp)[(((size_t)b * HH + h) * SS + s) * DEPTH + d] = f2bf(x);
      } else {
        ((float*)outp)[(size_t)m * N + n] = x;
      }
    }
  }
}

// ---------------------------------------------------------------------------
// Flash attention over split-head bf16 Q/K/V [B*H, S, 64].
// Block: 128 threads (4 waves) handles 64 q-rows of one (b,h).
// K tiles double-buffered via GLOBAL_LOAD_ASYNC_TO_LDS_B128 (ASYNCcnt),
// overlapping the next tile's HBM/L2 fetch with current tile's 16 WMMAs.
// Per-wave ASYNCcnt is in-order: wait<=4 releases everything but the 4
// chunks issued for the next tile.
// ---------------------------------------------------------------------------
__global__ __launch_bounds__(128) void attn_kernel(
    const __bf16* __restrict__ qh, const __bf16* __restrict__ kh,
    const __bf16* __restrict__ vh, const int* __restrict__ mask,
    __bf16* __restrict__ ctx) {
  constexpr int QT = 64, KTile = 64, ST = 72;  // stride 72: 16B-aligned frags
  __shared__ __align__(16) __bf16 Qs[QT * ST];
  __shared__ __align__(16) __bf16 Ks[2][KTile * ST];  // [key][depth], 2 bufs
  __shared__ __align__(16) __bf16 Vt[DEPTH * ST];     // [depth][key] (transposed)
  __shared__ __align__(16) __bf16 Ps[QT * ST];        // per-wave P staging

  const int tid  = threadIdx.x;
  const int wave = tid >> 5;
  const int lane = tid & 31;
  const int nIdx = lane & 15, hf = lane >> 4;
  const int bh = blockIdx.y;
  const int b = bh >> 4, h = bh & 15;
  const int q0 = blockIdx.x * QT;
  const int qm0 = wave * 16;

  const __bf16* Qg = qh + ((size_t)bh * SS + q0) * DEPTH;
  const __bf16* Kg = kh + (size_t)bh * SS * DEPTH;
  const __bf16* Vg = vh + (size_t)bh * SS * DEPTH;
  const int*  mrow = mask + (size_t)b * SS * SS;

  // Async-stage Q tile (4 chunks/thread) and K tile 0 (4 chunks/thread).
  for (int i = tid; i < QT * 8; i += 128) {
    int r = i >> 3, c = (i & 7) * 8;
    async_b128(&Qs[r * ST + c], Qg + (size_t)r * DEPTH + c);
  }
  for (int i = tid; i < KTile * 8; i += 128) {
    int r = i >> 3, c = (i & 7) * 8;
    async_b128(&Ks[0][r * ST + c], Kg + (size_t)r * DEPTH + c);
  }

  float rowM[8], rowL[8];
  v8f Oacc[4];
  for (int r = 0; r < 8; r++) { rowM[r] = -1e30f; rowL[r] = 0.f; }
  for (int t = 0; t < 4; t++)
    for (int e = 0; e < 8; e++) Oacc[t][e] = 0.f;
  const float scale = 0.125f;  // 1/sqrt(depth)

  for (int kb = 0; kb < SS; kb += KTile) {
    const int buf = (kb / KTile) & 1;
    __syncthreads();  // prior readers of Ks[buf] / Vt done

    // Kick off async copy of the *next* K tile into the other buffer.
    const bool haveNext = (kb + KTile) < SS;
    if (haveNext) {
      const __bf16* Kn = Kg + (size_t)(kb + KTile) * DEPTH;
      for (int i = tid; i < KTile * 8; i += 128) {
        int r = i >> 3, c = (i & 7) * 8;
        async_b128(&Ks[buf ^ 1][r * ST + c], Kn + (size_t)r * DEPTH + c);
      }
    }
    // Manual transposed staging of V (needs register transpose anyway).
    for (int i = tid; i < KTile * DEPTH; i += 128) {
      int r = i >> 6, c = i & 63;
      Vt[c * ST + r] = Vg[(size_t)(kb + r) * DEPTH + c];
    }
    // Release this tile's async chunks (keep next tile's 4 in flight).
    if (haveNext) WAIT_ASYNC(4);
    else          WAIT_ASYNC(0);
    __syncthreads();

    // ---- S = Q K^T (contraction over depth, K rows are depth-contiguous) --
    v8f Sacc[4];
    for (int t = 0; t < 4; t++)
      for (int e = 0; e < 8; e++) Sacc[t][e] = 0.f;
    for (int ds = 0; ds < DEPTH; ds += 32) {
      v16bf aq = load_frag_a(Qs, qm0, ST, ds);
      for (int t = 0; t < 4; t++) {
        v16bf bk = load_frag_b(Ks[buf], t * 16, ST, ds);
        Sacc[t] = __builtin_amdgcn_wmma_f32_16x16x32_bf16(
            false, aq, false, bk, (short)0, Sacc[t], false, false);
      }
    }

    // ---- scale + additive mask + per-row max (across 16 lanes of a half) --
    float tmax[8];
    for (int r = 0; r < 8; r++) tmax[r] = -1e30f;
    for (int t = 0; t < 4; t++) {
      int kcol = kb + t * 16 + nIdx;
      for (int r = 0; r < 8; r++) {
        int q = q0 + qm0 + hf * 8 + r;
        float mv = (float)mrow[(size_t)q * SS + kcol];
        float s = Sacc[t][r] * scale + mv * -1e9f;
        Sacc[t][r] = s;
        tmax[r] = fmaxf(tmax[r], s);
      }
    }
    for (int r = 0; r < 8; r++) {
      float v = tmax[r];
      v = fmaxf(v, __shfl_xor(v, 1, 16));
      v = fmaxf(v, __shfl_xor(v, 2, 16));
      v = fmaxf(v, __shfl_xor(v, 4, 16));
      v = fmaxf(v, __shfl_xor(v, 8, 16));
      tmax[r] = v;
    }

    // ---- online-softmax rescale ----
    float rsum[8];
    for (int r = 0; r < 8; r++) {
      float newM = fmaxf(rowM[r], tmax[r]);
      float corr = __expf(rowM[r] - newM);
      rowM[r] = newM;
      rowL[r] *= corr;
      for (int t = 0; t < 4; t++) Oacc[t][r] *= corr;
      rsum[r] = 0.f;
    }

    // ---- P = exp(S - m), bf16, staged to per-wave LDS for A-relayout ----
    __bf16* Pw = Ps + qm0 * ST;
    for (int t = 0; t < 4; t++) {
      for (int r = 0; r < 8; r++) {
        float p = __expf(Sacc[t][r] - rowM[r]);
        rsum[r] += p;
        Pw[(hf * 8 + r) * ST + t * 16 + nIdx] = f2bf(p);
      }
    }
    for (int r = 0; r < 8; r++) {
      float v = rsum[r];
      v += __shfl_xor(v, 1, 16);
      v += __shfl_xor(v, 2, 16);
      v += __shfl_xor(v, 4, 16);
      v += __shfl_xor(v, 8, 16);
      rowL[r] += v;
    }
    asm volatile("s_wait_dscnt 0" ::: "memory");  // Pw writes -> frag reads

    // ---- O += P @ V (contraction over keys; Vt[d][key] is key-contiguous) -
    for (int ks = 0; ks < KTile; ks += 32) {
      v16bf ap = load_frag_a(Pw, 0, ST, ks);
      for (int t = 0; t < 4; t++) {
        v16bf bv = load_frag_b(Vt, t * 16, ST, ks);
        Oacc[t] = __builtin_amdgcn_wmma_f32_16x16x32_bf16(
            false, ap, false, bv, (short)0, Oacc[t], false, false);
      }
    }
  }

  // ---- normalize + store ctx as [B,S,D] bf16 (heads re-interleaved) ----
  for (int t = 0; t < 4; t++) {
    int d = t * 16 + nIdx;
    for (int r = 0; r < 8; r++) {
      int q = q0 + qm0 + hf * 8 + r;
      float o = Oacc[t][r] / rowL[r];
      ctx[((size_t)b * SS + q) * DDIM + h * DEPTH + d] = f2bf(o);
    }
  }
}

// ---------------------------------------------------------------------------
extern "C" void kernel_launch(void* const* d_in, const int* in_sizes, int n_in,
                              void* d_out, int out_size, void* d_ws, size_t ws_size,
                              hipStream_t stream) {
  const float* value = (const float*)d_in[0];
  const float* key   = (const float*)d_in[1];
  const float* query = (const float*)d_in[2];
  const int*   mask  = (const int*)d_in[3];
  const float* Wq = (const float*)d_in[4];
  const float* bq = (const float*)d_in[5];
  const float* Wk = (const float*)d_in[6];
  const float* bk = (const float*)d_in[7];
  const float* Wv = (const float*)d_in[8];
  const float* bv = (const float*)d_in[9];
  const float* Wo = (const float*)d_in[10];
  const float* bo = (const float*)d_in[11];

  const size_t Mrows = (size_t)BB * SS;     // 8192
  const size_t elems = Mrows * DDIM;        // 8.39M bf16 per buffer
  __bf16* qhp = (__bf16*)d_ws;              // [B,H,S,64]
  __bf16* khp = qhp + elems;
  __bf16* vhp = khp + elems;
  __bf16* ctx = vhp + elems;                // [B,S,D]

  dim3 gG((unsigned)(Mrows / 128), DDIM / 128);
  gemm_elu_kernel<false, true><<<gG, 256, 0, stream>>>(query, Wq, bq, qhp,
                                                       (int)Mrows, DDIM, DDIM);
  gemm_elu_kernel<false, true><<<gG, 256, 0, stream>>>(key,   Wk, bk, khp,
                                                       (int)Mrows, DDIM, DDIM);
  gemm_elu_kernel<false, true><<<gG, 256, 0, stream>>>(value, Wv, bv, vhp,
                                                       (int)Mrows, DDIM, DDIM);

  dim3 gA(SS / 64, BB * HH);
  attn_kernel<<<gA, 128, 0, stream>>>(qhp, khp, vhp, mask, ctx);

  gemm_elu_kernel<true, false><<<gG, 256, 0, stream>>>(ctx, Wo, bo, d_out,
                                                       (int)Mrows, DDIM, DDIM);
}
